// MultiheadAttention_39711267619126
// MI455X (gfx1250) — matrix operand order repaired and checked by reading, not compiled, not run on hardware
//
#include <hip/hip_runtime.h>

// ---------------------------------------------------------------------------
// MultiheadAttention forward for MI455X (gfx1250), bf16 WMMA pipeline.
//   B=4, S=2048, D=1024, H=16, dk=64
// Stages:
//   1) transpose+convert weights to bf16 (B-operand wants columns contiguous)
//   2) Q/K/V projection GEMMs: W panel async-staged to LDS (double buffered),
//      fp32 A converted inline with native bf16 cvt, f32 accumulate
//   3) flash attention: async-copy K/V chunks to LDS (double buffered,
//      ASYNCcnt), WMMA QK^T + PV, DPP row_ror butterfly softmax reductions
//   4) output GEMM -> fp32 d_out
// ---------------------------------------------------------------------------

#define D_MODEL 1024
#define SEQ     2048
#define BATCH   4
#define HEADS   16
#define DK      64
#define ROWS    (BATCH * SEQ)   // 8192
#define NCHUNKS (SEQ / 64)      // 32
#define KCHUNKS (D_MODEL / 128) // 8

typedef __attribute__((ext_vector_type(16))) __bf16 v16bf;
typedef __attribute__((ext_vector_type(8)))  __bf16 v8bf;
typedef __attribute__((ext_vector_type(8)))  float  v8f;
typedef __attribute__((ext_vector_type(4)))  int    v4i;

__device__ __forceinline__ __bf16 f2bf(float f) {
    return (__bf16)f;    // native v_cvt (RNE) on gfx1250
}

__device__ __forceinline__ v8f wmma_bf16(v16bf a, v16bf b, v8f c) {
    return __builtin_amdgcn_wmma_f32_16x16x32_bf16(
        false, a, false, b, (short)0, c, false, false);
}

// ---------------- cross-lane reductions within a 16-lane half -------------
#if __has_builtin(__builtin_amdgcn_update_dpp)
template <int CTRL>
__device__ __forceinline__ float dpp_ror(float x) {
    return __builtin_bit_cast(float,
        __builtin_amdgcn_update_dpp(0, __builtin_bit_cast(int, x),
                                    CTRL, 0xF, 0xF, true));
}
__device__ __forceinline__ float rowmax16(float x) {
    x = fmaxf(x, dpp_ror<0x121>(x));   // row_ror:1
    x = fmaxf(x, dpp_ror<0x122>(x));   // row_ror:2
    x = fmaxf(x, dpp_ror<0x124>(x));   // row_ror:4
    x = fmaxf(x, dpp_ror<0x128>(x));   // row_ror:8
    return x;
}
__device__ __forceinline__ float rowsum16(float x) {
    x += dpp_ror<0x121>(x);
    x += dpp_ror<0x122>(x);
    x += dpp_ror<0x124>(x);
    x += dpp_ror<0x128>(x);
    return x;
}
#else
__device__ __forceinline__ float rowmax16(float x) {
    x = fmaxf(x, __shfl_xor(x, 1, 32));
    x = fmaxf(x, __shfl_xor(x, 2, 32));
    x = fmaxf(x, __shfl_xor(x, 4, 32));
    x = fmaxf(x, __shfl_xor(x, 8, 32));
    return x;
}
__device__ __forceinline__ float rowsum16(float x) {
    x += __shfl_xor(x, 1, 32);
    x += __shfl_xor(x, 2, 32);
    x += __shfl_xor(x, 4, 32);
    x += __shfl_xor(x, 8, 32);
    return x;
}
#endif

// ---------------- async global->LDS staging (gfx1250 path) ----------------
#if defined(__AMDGCN__) && \
    __has_builtin(__builtin_amdgcn_global_load_async_to_lds_b128) && \
    __has_builtin(__builtin_amdgcn_s_wait_asynccnt)
#define ASYNC_LDS 1
#else
#define ASYNC_LDS 0
#endif

#if ASYNC_LDS
typedef __attribute__((address_space(1))) v4i* gp_v4i;   // global int4*
typedef __attribute__((address_space(3))) v4i* lp_v4i;   // LDS int4*
__device__ __forceinline__ void cp16_g2l(const __bf16* gsrc, __bf16* ldst) {
    __builtin_amdgcn_global_load_async_to_lds_b128(
        (gp_v4i)(void*)const_cast<__bf16*>(gsrc),
        (lp_v4i)(void*)ldst, 0, 0);
}
#define STAGE_WAIT_PREV() __builtin_amdgcn_s_wait_asynccnt(4)
#define STAGE_WAIT_ALL()  __builtin_amdgcn_s_wait_asynccnt(0)
#else
__device__ __forceinline__ void cp16_g2l(const __bf16* gsrc, __bf16* ldst) {
    *(v8bf*)ldst = *(const v8bf*)gsrc;       // register bounce fallback
}
#define STAGE_WAIT_PREV() ((void)0)
#define STAGE_WAIT_ALL()  ((void)0)
#endif

// Stage one 64-key chunk: K rows [t0,t0+64) x 64dk and V^T 64dk x [t0,t0+64).
// 4 x b128 per thread (2 K + 2 V), 256 threads -> 16 KB.
__device__ __forceinline__ void stage_chunk(const __bf16* __restrict__ Kh,
                                            const __bf16* __restrict__ Vh,
                                            __bf16* __restrict__ kbuf,
                                            __bf16* __restrict__ vbuf,
                                            int t0, int tid) {
#pragma unroll
    for (int rep = 0; rep < 2; ++rep) {
        const int lin = tid + rep * 256;          // 0..511
        const int row = lin >> 3;                 // 0..63
        const int c8  = lin & 7;                  // 16B column
        cp16_g2l(Kh + (size_t)(t0 + row) * DK + c8 * 8,  kbuf + row * 64 + c8 * 8);
        cp16_g2l(Vh + (size_t)row * SEQ + t0 + c8 * 8,   vbuf + row * 64 + c8 * 8);
    }
}

// Stage a 64n x 128k W panel (16 KB): 4 x b128 per thread.
__device__ __forceinline__ void stage_w(const __bf16* __restrict__ Wt,
                                        __bf16* __restrict__ wbuf,
                                        int n_base, int k0, int tid) {
#pragma unroll
    for (int rep = 0; rep < 4; ++rep) {
        const int lin = tid + rep * 256;          // 0..1023
        const int nlcl = lin >> 4;                // 0..63
        const int c8   = lin & 15;                // 16 x 16B = 128 elements
        cp16_g2l(Wt + (size_t)(n_base + nlcl) * D_MODEL + k0 + c8 * 8,
                 wbuf + nlcl * 128 + c8 * 8);
    }
}

// ---------------------------------------------------------------------------
// Weight transpose + fp32->bf16:  Wt[n][k] = (bf16) W[k][n]
// ---------------------------------------------------------------------------
__global__ void wt_kernel(const float* __restrict__ W, __bf16* __restrict__ Wt) {
    __shared__ __bf16 tile[32][33];
    const int tx = threadIdx.x, ty = threadIdx.y;     // 32 x 8
    const int n0 = blockIdx.x * 32, k0 = blockIdx.y * 32;
#pragma unroll
    for (int i = 0; i < 32; i += 8)
        tile[ty + i][tx] = f2bf(W[(size_t)(k0 + ty + i) * D_MODEL + n0 + tx]);
    __syncthreads();
#pragma unroll
    for (int i = 0; i < 32; i += 8)
        Wt[(size_t)(n0 + ty + i) * D_MODEL + k0 + tx] = tile[tx][ty + i];
}

// ---------------------------------------------------------------------------
// GEMM: C[8192 x 1024] = A[8192 x 1024] * W[1024 x 1024] + bias
// W panel staged to LDS (async, double buffered); 4 k-steps per staged panel.
//  MODE 0: A fp32, out bf16 scattered as [B,H,S,dk]        (Q, K projections)
//  MODE 1: A fp32, out bf16 scattered as [B,H,dk,S]        (V projection)
//  MODE 2: A bf16 row-major, out fp32 row-major            (output projection)
// ---------------------------------------------------------------------------
template <int MODE>
__global__ void gemm_kernel(const float*  __restrict__ Af32,
                            const __bf16* __restrict__ Abf,
                            const __bf16* __restrict__ Wt,   // [n][k] bf16
                            const float*  __restrict__ bias,
                            void*         __restrict__ out) {
    __shared__ __align__(16) __bf16 wbuf[2][64][128];   // 32 KB

    const int tid  = threadIdx.x;
    const int wv   = tid >> 5;
    const int lane = tid & 31;
    const int nl   = lane & 15;
    const int half = lane >> 4;
    const int m_base = blockIdx.y * 128 + wv * 16;
    const int n_base = blockIdx.x * 64;
    const int m = m_base + nl;

    v8f acc[4] = {};

    stage_w(Wt, &wbuf[0][0][0], n_base, 0, tid);

    for (int kc = 0; kc < KCHUNKS; ++kc) {
        const int cur = kc & 1;
        if (kc + 1 < KCHUNKS) {
            stage_w(Wt, &wbuf[cur ^ 1][0][0], n_base, (kc + 1) * 128, tid);
            STAGE_WAIT_PREV();
        } else {
            STAGE_WAIT_ALL();
        }
        __syncthreads();

#pragma unroll
        for (int kk = 0; kk < 4; ++kk) {
            const int k0 = kc * 128 + kk * 32;
            // ---- A fragment: 16x32, lane = M, K chunk [h*16,+16)
            v16bf a;
            if constexpr (MODE == 2) {
                const __bf16* ap = Abf + (size_t)m * D_MODEL + k0 + half * 16;
                v8bf lo = *(const v8bf*)ap;
                v8bf hi = *(const v8bf*)(ap + 8);
#pragma unroll
                for (int i = 0; i < 8; ++i) { a[i] = lo[i]; a[i + 8] = hi[i]; }
            } else {
                const float* xp = Af32 + (size_t)m * D_MODEL + k0 + half * 16;
#pragma unroll
                for (int i = 0; i < 8; ++i) a[i]     = f2bf(xp[i]);
#pragma unroll
                for (int i = 0; i < 8; ++i) a[i + 8] = f2bf(xp[8 + i]);
            }
            // ---- B fragments from LDS panel
#pragma unroll
            for (int j = 0; j < 4; ++j) {
                const __bf16* bp = &wbuf[cur][j * 16 + nl][kk * 32 + half * 16];
                v16bf bfrag = *(const v16bf*)bp;
                acc[j] = wmma_bf16(a, bfrag, acc[j]);
            }
        }
        __syncthreads();
    }

#pragma unroll
    for (int j = 0; j < 4; ++j) {
        const int n  = n_base + j * 16 + nl;
        const float bv = bias[n];
#pragma unroll
        for (int r = 0; r < 8; ++r) {
            const float v  = acc[j][r] + bv;
            const int  row = m_base + half * 8 + r;
            if constexpr (MODE == 2) {
                ((float*)out)[(size_t)row * D_MODEL + n] = v;
            } else {
                const int b = row >> 11;
                const int s = row & (SEQ - 1);
                const int h = n >> 6;
                const int d = n & (DK - 1);
                size_t idx;
                if constexpr (MODE == 0)
                    idx = (((size_t)(b * HEADS + h)) * SEQ + s) * DK + d;
                else
                    idx = (((size_t)(b * HEADS + h)) * DK + d) * SEQ + s;
                ((__bf16*)out)[idx] = f2bf(v);
            }
        }
    }
}

// ---------------------------------------------------------------------------
// Flash attention. grid = (SEQ/128, B*H), block = 256 (8 waves).
// K/V staged to LDS in 64-key chunks, double buffered with async copies.
// Wave owns 16 query rows; online softmax with DPP row reductions.
// ---------------------------------------------------------------------------
__global__ void attn_kernel(const __bf16* __restrict__ Qp,   // [B,H,S,dk]
                            const __bf16* __restrict__ Kp,   // [B,H,S,dk]
                            const __bf16* __restrict__ Vt,   // [B,H,dk,S]
                            __bf16*       __restrict__ Ar) { // [B,S,D] permuted
    __shared__ __align__(16) __bf16 kbuf[2][64][64];   // [t][d]
    __shared__ __align__(16) __bf16 vbuf[2][64][64];   // [d][t]
    __shared__ __align__(16) __bf16 pshm[8][16][64];

    const int bh = blockIdx.y;
    const int b  = bh >> 4, h = bh & (HEADS - 1);
    const int tid  = threadIdx.x;
    const int wv   = tid >> 5;
    const int lane = tid & 31;
    const int nl   = lane & 15;
    const int half = lane >> 4;

    const __bf16* Qh = Qp + (size_t)bh * SEQ * DK;
    const __bf16* Kh = Kp + (size_t)bh * SEQ * DK;
    const __bf16* Vh = Vt + (size_t)bh * DK * SEQ;

    const int m_base = blockIdx.x * 128 + wv * 16;

    // Q A-fragments, resident for the whole kernel (dk=64 -> 2 k-steps)
    const __bf16* qrow = Qh + (size_t)(m_base + nl) * DK;
    v16bf qa[2];
#pragma unroll
    for (int ks = 0; ks < 2; ++ks) {
        v8bf lo = *(const v8bf*)(qrow + ks * 32 + half * 16);
        v8bf hi = *(const v8bf*)(qrow + ks * 32 + half * 16 + 8);
#pragma unroll
        for (int i = 0; i < 8; ++i) { qa[ks][i] = lo[i]; qa[ks][i + 8] = hi[i]; }
    }

    v8f o[4] = {};
    float ms[8], ls[8];
#pragma unroll
    for (int r = 0; r < 8; ++r) { ms[r] = -3.0e38f; ls[r] = 0.0f; }

    stage_chunk(Kh, Vh, &kbuf[0][0][0], &vbuf[0][0][0], 0, tid);

    for (int c = 0; c < NCHUNKS; ++c) {
        const int cur = c & 1;
        if (c + 1 < NCHUNKS) {
            stage_chunk(Kh, Vh, &kbuf[cur ^ 1][0][0], &vbuf[cur ^ 1][0][0],
                        (c + 1) * 64, tid);
            STAGE_WAIT_PREV();            // allow next chunk's 4 in flight
        } else {
            STAGE_WAIT_ALL();
        }
        __syncthreads();                  // chunk c visible to all waves

        // ---- scores: 16x64 = Q(16x64) * K_chunk^T(64x64)
        v8f sc[4];
#pragma unroll
        for (int jn = 0; jn < 4; ++jn) {
            const __bf16* kr = &kbuf[cur][jn * 16 + nl][0];
            v16bf k0 = *(const v16bf*)(kr + half * 16);
            v16bf k1 = *(const v16bf*)(kr + 32 + half * 16);
            v8f z = {};
            z = wmma_bf16(qa[0], k0, z);
            z = wmma_bf16(qa[1], k1, z);
            sc[jn] = z;
        }

        // ---- online softmax, DPP butterfly reductions per 16-lane half
#pragma unroll
        for (int r = 0; r < 8; ++r) {
            const float s0 = sc[0][r] * 0.125f;   // 1/sqrt(64)
            const float s1 = sc[1][r] * 0.125f;
            const float s2 = sc[2][r] * 0.125f;
            const float s3 = sc[3][r] * 0.125f;
            float mx = rowmax16(fmaxf(fmaxf(s0, s1), fmaxf(s2, s3)));
            const float mnew  = fmaxf(ms[r], mx);
            const float alpha = __expf(ms[r] - mnew);
            const float p0 = __expf(s0 - mnew);
            const float p1 = __expf(s1 - mnew);
            const float p2 = __expf(s2 - mnew);
            const float p3 = __expf(s3 - mnew);
            const float rs = rowsum16((p0 + p1) + (p2 + p3));
            ls[r] = ls[r] * alpha + rs;
            ms[r] = mnew;
#pragma unroll
            for (int j = 0; j < 4; ++j) o[j][r] *= alpha;
            const int prow = half * 8 + r;
            pshm[wv][prow][nl]      = f2bf(p0);
            pshm[wv][prow][16 + nl] = f2bf(p1);
            pshm[wv][prow][32 + nl] = f2bf(p2);
            pshm[wv][prow][48 + nl] = f2bf(p3);
        }
        __builtin_amdgcn_wave_barrier();
        // ---- reload P (16x64) as two A fragments
        v16bf pa[2];
#pragma unroll
        for (int ks = 0; ks < 2; ++ks) {
            v8bf lo = *(const v8bf*)&pshm[wv][nl][ks * 32 + half * 16];
            v8bf hi = *(const v8bf*)&pshm[wv][nl][ks * 32 + half * 16 + 8];
#pragma unroll
            for (int i = 0; i < 8; ++i) { pa[ks][i] = lo[i]; pa[ks][i + 8] = hi[i]; }
        }
        __builtin_amdgcn_wave_barrier();
        // ---- O += P(16x64) * V_chunk(64x64)
#pragma unroll
        for (int j = 0; j < 4; ++j) {
            const __bf16* vr = &vbuf[cur][j * 16 + nl][0];
            v16bf v0 = *(const v16bf*)(vr + half * 16);
            v16bf v1 = *(const v16bf*)(vr + 32 + half * 16);
            o[j] = wmma_bf16(pa[0], v0, o[j]);
            o[j] = wmma_bf16(pa[1], v1, o[j]);
        }
        __syncthreads();                  // done reading before re-stage
    }

    // ---- normalize + store in permuted reshape order (bf16 A for out-GEMM)
#pragma unroll
    for (int r = 0; r < 8; ++r) {
        const float inv = 1.0f / ls[r];
        const int srow  = m_base + half * 8 + r;
        const size_t base =
            ((size_t)b * SEQ + h * 128 + (srow >> 4)) * D_MODEL +
            (size_t)(srow & 15) * DK;
#pragma unroll
        for (int j = 0; j < 4; ++j)
            Ar[base + j * 16 + nl] = f2bf(o[j][r] * inv);
    }
}

// ---------------------------------------------------------------------------
extern "C" void kernel_launch(void* const* d_in, const int* in_sizes, int n_in,
                              void* d_out, int out_size, void* d_ws, size_t ws_size,
                              hipStream_t stream) {
    (void)in_sizes; (void)n_in; (void)out_size; (void)ws_size;

    const float* q   = (const float*)d_in[0];
    const float* k   = (const float*)d_in[1];
    const float* v   = (const float*)d_in[2];
    const float* w_q = (const float*)d_in[3];
    const float* b_q = (const float*)d_in[4];
    const float* w_k = (const float*)d_in[5];
    const float* b_k = (const float*)d_in[6];
    const float* w_v = (const float*)d_in[7];
    const float* b_v = (const float*)d_in[8];
    const float* w_0 = (const float*)d_in[9];
    const float* b_0 = (const float*)d_in[10];

    // workspace layout (72 MB total)
    __bf16* WtQ = (__bf16*)d_ws;
    __bf16* WtK = WtQ + (size_t)D_MODEL * D_MODEL;
    __bf16* WtV = WtK + (size_t)D_MODEL * D_MODEL;
    __bf16* WtO = WtV + (size_t)D_MODEL * D_MODEL;
    __bf16* Qp  = WtO + (size_t)D_MODEL * D_MODEL;
    __bf16* Kp  = Qp  + (size_t)ROWS * D_MODEL;
    __bf16* Vt  = Kp  + (size_t)ROWS * D_MODEL;
    __bf16* Ar  = Vt  + (size_t)ROWS * D_MODEL;

    {
        dim3 g(D_MODEL / 32, D_MODEL / 32), t(32, 8);
        wt_kernel<<<g, t, 0, stream>>>(w_q, WtQ);
        wt_kernel<<<g, t, 0, stream>>>(w_k, WtK);
        wt_kernel<<<g, t, 0, stream>>>(w_v, WtV);
        wt_kernel<<<g, t, 0, stream>>>(w_0, WtO);
    }
    {
        dim3 g(D_MODEL / 64, ROWS / 128);
        gemm_kernel<0><<<g, 256, 0, stream>>>(q, nullptr, WtQ, b_q, Qp);
        gemm_kernel<0><<<g, 256, 0, stream>>>(k, nullptr, WtK, b_k, Kp);
        gemm_kernel<1><<<g, 256, 0, stream>>>(v, nullptr, WtV, b_v, Vt);
    }
    attn_kernel<<<dim3(SEQ / 128, BATCH * HEADS), 256, 0, stream>>>(Qp, Kp, Vt, Ar);
    gemm_kernel<2><<<dim3(D_MODEL / 64, ROWS / 128), 256, 0, stream>>>(
        nullptr, Ar, WtO, b_0, (float*)d_out);
}
